// DeepRecurrentGeneration_21431886807311
// MI455X (gfx1250) — matrix operand order repaired and checked by reading, not compiled, not run on hardware
//
#include <hip/hip_runtime.h>
#include <hip/hip_bf16.h>

// Problem constants
static constexpr int HID  = 256;   // hidden
static constexpr int LAT  = 256;   // latent
static constexpr int BAT  = 1024;  // batch
static constexpr int DAYS = 128;   // time steps
static constexpr float KEEP = 0.6f;
static constexpr int ROWS_PER_BLK = 32;            // two 16-row WMMA M tiles
static constexpr int NBLK = BAT / ROWS_PER_BLK;    // 32 blocks

typedef __attribute__((ext_vector_type(16))) __bf16 bf16x16;
typedef __attribute__((ext_vector_type(8)))  __bf16 bf16x8;
typedef __attribute__((ext_vector_type(8)))  float  f32x8;

#define ZERO8 {0.f,0.f,0.f,0.f,0.f,0.f,0.f,0.f}

__device__ __forceinline__ f32x8 wmma_bf16(bf16x16 a, bf16x16 b, f32x8 c) {
  // D = A(16x32 bf16) * B(32x16 bf16) + C(16x16 f32)
  return __builtin_amdgcn_wmma_f32_16x16x32_bf16(false, a, false, b, (short)0, c, false, false);
}

// A fragment from a row-major 16x256 bf16 tile in LDS.
// lane (hf,lq): row = lq; chunk0 = K[kt*32 + hf*8 .. +7], chunk1 = K[kt*32+16+hf*8 .. +7]
__device__ __forceinline__ bf16x16 loadA(const __bf16* tile, int kt, int lq, int hf) {
  const __bf16* p = tile + lq * 256 + kt * 32 + hf * 8;
  bf16x8 c0 = *(const bf16x8*)p;
  bf16x8 c1 = *(const bf16x8*)(p + 16);
  return __builtin_shufflevector(c0, c1, 0,1,2,3,4,5,6,7,8,9,10,11,12,13,14,15);
}

__device__ __forceinline__ float sigm(float x) { return 1.0f / (1.0f + __expf(-x)); }
__device__ __forceinline__ float tanh_f(float x) { return 2.0f * sigm(2.0f * x) - 1.0f; }

// ---------------------------------------------------------------------------
// Pack f32 weight W[N][K] (row-major) into per-lane-contiguous B fragments:
// dst[((nT*KT + kt)*32 + lane)*16 + j] = bf16(W[nT*16 + (lane&15)][kt*32 + (lane>>4)*8 + j + (j>=8 ? 8 : 0)])
// ---------------------------------------------------------------------------
__global__ void pack_w(const float* __restrict__ W, __bf16* __restrict__ dst,
                       int KT, int K, int total) {
  int id = blockIdx.x * blockDim.x + threadIdx.x;
  if (id >= total) return;
  int lane = id & 31;
  int kt   = (id >> 5) % KT;
  int nT   = id / (32 * KT);
  int lq = lane & 15, hf = lane >> 4;
  const float* src = W + (size_t)(nT * 16 + lq) * K + kt * 32 + hf * 8;
  __bf16* d = dst + (size_t)id * 16;
#pragma unroll
  for (int j = 0; j < 16; ++j) d[j] = (__bf16)src[j + (j >= 8 ? 8 : 0)];
}

// ---------------------------------------------------------------------------
// Main persistent recurrent kernel.
// One block = 32 batch rows (two 16-row M tiles), full T loop.
// 4 waves split the N (gate-column) dimension; each B fragment is reused for
// both M tiles (halves weight L2 traffic per FLOP). State lives in LDS (bf16).
// Dynamic LDS: hA 2x4096 | encH 2p x 2m x 4096 | zA 2p x 2m x 4096 | reduce bufs
// ---------------------------------------------------------------------------
__global__ __launch_bounds__(128) void drg_main(
    const float* __restrict__ hseries, const float* __restrict__ eps,
    const float* __restrict__ dropu,
    const float* __restrict__ enc_b_ih, const float* __restrict__ enc_b_hh,
    const float* __restrict__ dec_b_ih, const float* __restrict__ dec_b_hh,
    const float* __restrict__ b_mu, const float* __restrict__ b_ls,
    const float* __restrict__ w_pred, const float* __restrict__ b_pred,
    const __bf16* __restrict__ pw_enc_ih, const __bf16* __restrict__ pw_enc_hh,
    const __bf16* __restrict__ pw_mu, const __bf16* __restrict__ pw_ls,
    const __bf16* __restrict__ pw_dec,
    float* __restrict__ out_pred, float* __restrict__ out_z,
    float* __restrict__ klPart)
{
  extern __shared__ __align__(32) __bf16 smem[];
  __bf16* hA   = smem;                 // [2m][4096]
  __bf16* encH = smem + 2 * 4096;      // [2p][2m][4096]
  __bf16* zA   = smem + 6 * 4096;      // [2p][2m][4096]
  float* predPart = (float*)(smem + 10 * 4096);   // [4 waves][32 rows]
  float* klS      = predPart + 4 * 32;            // [4]

  const int tid  = threadIdx.x;
  const int lane = tid & 31;
  const int wave = tid >> 5;
  const int lq   = lane & 15;
  const int hf   = lane >> 4;
  const int mBase = blockIdx.x * ROWS_PER_BLK;

  for (int i = tid; i < 2 * 4096; i += 128) {
    encH[i] = (__bf16)0.f;   // p=0 slice
    zA[i]   = (__bf16)0.f;   // p=0 slice
  }
  float klAcc = 0.f;
  int p = 0;
  __syncthreads();

  for (int t = 0; t < DAYS; ++t) {
    const __bf16* encHp = encH + p * 8192;         // old enc hidden
    __bf16*       encHn = encH + (p ^ 1) * 8192;   // new enc hidden
    const __bf16* zAp   = zA + p * 8192;           // z_{t-1}
    __bf16*       zAn   = zA + (p ^ 1) * 8192;     // z_t

    // ---- phase 0: stage h_t tile (f32 -> bf16, row-major, coalesced) ----
    for (int i4 = tid; i4 < (ROWS_PER_BLK * 256) / 4; i4 += 128) {
      int e = i4 * 4;
      int row = e >> 8, k = e & 255;
      const float4 v = *(const float4*)&hseries[((size_t)(mBase + row) * DAYS + t) * HID + k];
      __bf16* d = &hA[(row >> 4) * 4096 + (row & 15) * 256 + k];
      d[0] = (__bf16)v.x; d[1] = (__bf16)v.y; d[2] = (__bf16)v.z; d[3] = (__bf16)v.w;
    }
    __syncthreads();

    // ---- phase 1: encoder GRU cell, gates fused into WMMA epilogue ----
    for (int nT = wave; nT < 16; nT += 4) {
      f32x8 air[2] = {ZERO8, ZERO8}, aiz[2] = {ZERO8, ZERO8}, ain[2] = {ZERO8, ZERO8};
      f32x8 ahr[2] = {ZERO8, ZERO8}, ahz[2] = {ZERO8, ZERO8}, ahn[2] = {ZERO8, ZERO8};
      const bf16x16* wih = (const bf16x16*)pw_enc_ih;
      const bf16x16* whh = (const bf16x16*)pw_enc_hh;
#pragma unroll 1
      for (int kt = 0; kt < 8; ++kt) {
        bf16x16 aH0 = loadA(hA,          kt, lq, hf);
        bf16x16 aH1 = loadA(hA + 4096,   kt, lq, hf);
        bf16x16 aP0 = loadA(encHp,        kt, lq, hf);
        bf16x16 aP1 = loadA(encHp + 4096, kt, lq, hf);
        bf16x16 bir = wih[((nT     ) * 8 + kt) * 32 + lane];
        bf16x16 biz = wih[((nT + 16) * 8 + kt) * 32 + lane];
        bf16x16 bin = wih[((nT + 32) * 8 + kt) * 32 + lane];
        bf16x16 bhr = whh[((nT     ) * 8 + kt) * 32 + lane];
        bf16x16 bhz = whh[((nT + 16) * 8 + kt) * 32 + lane];
        bf16x16 bhn = whh[((nT + 32) * 8 + kt) * 32 + lane];
        air[0] = wmma_bf16(aH0, bir, air[0]);  air[1] = wmma_bf16(aH1, bir, air[1]);
        aiz[0] = wmma_bf16(aH0, biz, aiz[0]);  aiz[1] = wmma_bf16(aH1, biz, aiz[1]);
        ain[0] = wmma_bf16(aH0, bin, ain[0]);  ain[1] = wmma_bf16(aH1, bin, ain[1]);
        ahr[0] = wmma_bf16(aP0, bhr, ahr[0]);  ahr[1] = wmma_bf16(aP1, bhr, ahr[1]);
        ahz[0] = wmma_bf16(aP0, bhz, ahz[0]);  ahz[1] = wmma_bf16(aP1, bhz, ahz[1]);
        ahn[0] = wmma_bf16(aP0, bhn, ahn[0]);  ahn[1] = wmma_bf16(aP1, bhn, ahn[1]);
      }
      int n = nT * 16 + lq;
      float bir_ = enc_b_ih[n], biz_ = enc_b_ih[HID + n], bin_ = enc_b_ih[2 * HID + n];
      float bhr_ = enc_b_hh[n], bhz_ = enc_b_hh[HID + n], bhn_ = enc_b_hh[2 * HID + n];
#pragma unroll
      for (int m = 0; m < 2; ++m) {
#pragma unroll
        for (int i = 0; i < 8; ++i) {
          int row = hf * 8 + i;
          float r  = sigm((air[m][i] + bir_) + (ahr[m][i] + bhr_));
          float zg = sigm((aiz[m][i] + biz_) + (ahz[m][i] + bhz_));
          float nn = tanh_f((ain[m][i] + bin_) + r * (ahn[m][i] + bhn_));
          float hp = (float)encHp[m * 4096 + row * 256 + n];
          encHn[m * 4096 + row * 256 + n] = (__bf16)((1.f - zg) * nn + zg * hp);
        }
      }
    }
    __syncthreads();

    // ---- phase 2: posterior heads (mu, log_sigma), reparam, KL ----
    for (int nT = wave; nT < 16; nT += 4) {
      f32x8 amu[2] = {ZERO8, ZERO8}, als[2] = {ZERO8, ZERO8};
      const bf16x16* wm = (const bf16x16*)pw_mu;
      const bf16x16* wl = (const bf16x16*)pw_ls;
#pragma unroll 2
      for (int kt = 0; kt < 16; ++kt) {
        const __bf16* src = (kt < 8) ? (encHn + kt * 32) : (zAp + (kt - 8) * 32);
        bf16x16 a0 = loadA(src,        0, lq, hf);
        bf16x16 a1 = loadA(src + 4096, 0, lq, hf);
        bf16x16 bm_ = wm[(nT * 16 + kt) * 32 + lane];
        bf16x16 bl_ = wl[(nT * 16 + kt) * 32 + lane];
        amu[0] = wmma_bf16(a0, bm_, amu[0]);  amu[1] = wmma_bf16(a1, bm_, amu[1]);
        als[0] = wmma_bf16(a0, bl_, als[0]);  als[1] = wmma_bf16(a1, bl_, als[1]);
      }
      int n = nT * 16 + lq;
      float bm = b_mu[n], bl = b_ls[n];
#pragma unroll
      for (int m = 0; m < 2; ++m) {
#pragma unroll
        for (int i = 0; i < 8; ++i) {
          int row = m * 16 + hf * 8 + i;
          size_t gidx = ((size_t)(mBase + row) * DAYS + t) * LAT + n;
          float mu = amu[m][i] + bm;
          float ls = als[m][i] + bl;
          float e  = __expf(ls);
          float zv = mu + e * eps[gidx];
          klAcc += (1.f + ls - mu * mu - e);
          out_z[gidx] = zv;
          zAn[m * 4096 + (hf * 8 + i) * 256 + n] = (__bf16)zv;
        }
      }
    }
    __syncthreads();

    // ---- phase 3: decoder GRU (h0==0 => gh = b_hh), dropout, pred head ----
    float predAcc[2][8];
#pragma unroll
    for (int m = 0; m < 2; ++m)
#pragma unroll
      for (int i = 0; i < 8; ++i) predAcc[m][i] = 0.f;
    for (int nT = wave; nT < 16; nT += 4) {
      f32x8 air[2] = {ZERO8, ZERO8}, aiz[2] = {ZERO8, ZERO8}, ain[2] = {ZERO8, ZERO8};
      const bf16x16* wd = (const bf16x16*)pw_dec;
#pragma unroll 2
      for (int kt = 0; kt < 16; ++kt) {
        const __bf16* src = (kt < 8) ? (hA + kt * 32) : (zAn + (kt - 8) * 32);
        bf16x16 a0 = loadA(src,        0, lq, hf);
        bf16x16 a1 = loadA(src + 4096, 0, lq, hf);
        bf16x16 br = wd[((nT     ) * 16 + kt) * 32 + lane];
        bf16x16 bz = wd[((nT + 16) * 16 + kt) * 32 + lane];
        bf16x16 bn = wd[((nT + 32) * 16 + kt) * 32 + lane];
        air[0] = wmma_bf16(a0, br, air[0]);  air[1] = wmma_bf16(a1, br, air[1]);
        aiz[0] = wmma_bf16(a0, bz, aiz[0]);  aiz[1] = wmma_bf16(a1, bz, aiz[1]);
        ain[0] = wmma_bf16(a0, bn, ain[0]);  ain[1] = wmma_bf16(a1, bn, ain[1]);
      }
      int n = nT * 16 + lq;
      float bir_ = dec_b_ih[n] + dec_b_hh[n];
      float biz_ = dec_b_ih[HID + n] + dec_b_hh[HID + n];
      float bin_ = dec_b_ih[2 * HID + n];
      float bhn_ = dec_b_hh[2 * HID + n];
      float wp   = w_pred[n];
#pragma unroll
      for (int m = 0; m < 2; ++m) {
#pragma unroll
        for (int i = 0; i < 8; ++i) {
          int row = m * 16 + hf * 8 + i;
          float r  = sigm(air[m][i] + bir_);
          float zg = sigm(aiz[m][i] + biz_);
          float nn = tanh_f(ain[m][i] + bin_ + r * bhn_);
          float dh = (1.f - zg) * nn;
          float u = dropu[((size_t)(mBase + row) * DAYS + t) * HID + n];
          dh = (u < KEEP) ? dh * (1.f / KEEP) : 0.f;
          predAcc[m][i] += dh * wp;
        }
      }
    }
    // reduce pred over the 16 lanes sharing each row-half
#pragma unroll
    for (int off = 1; off < 16; off <<= 1) {
#pragma unroll
      for (int m = 0; m < 2; ++m)
#pragma unroll
        for (int i = 0; i < 8; ++i)
          predAcc[m][i] += __shfl_xor(predAcc[m][i], off, 32);
    }
    if (lq == 0) {
#pragma unroll
      for (int m = 0; m < 2; ++m)
#pragma unroll
        for (int i = 0; i < 8; ++i)
          predPart[wave * 32 + m * 16 + hf * 8 + i] = predAcc[m][i];
    }
    __syncthreads();
    if (tid < ROWS_PER_BLK) {
      float s = predPart[tid] + predPart[32 + tid] + predPart[64 + tid] +
                predPart[96 + tid] + b_pred[0];
      out_pred[(size_t)(mBase + tid) * DAYS + t] = sigm(s);
    }
    __syncthreads();
    p ^= 1;
  }

  // deterministic KL partial per block
#pragma unroll
  for (int off = 1; off < 32; off <<= 1) klAcc += __shfl_xor(klAcc, off, 32);
  if (lane == 0) klS[wave] = klAcc;
  __syncthreads();
  if (tid == 0) klPart[blockIdx.x] = klS[0] + klS[1] + klS[2] + klS[3];
}

__global__ void kl_fin(const float* __restrict__ parts, float* __restrict__ out) {
  float s = 0.f;
  for (int i = 0; i < NBLK; ++i) s += parts[i];
  *out = (-0.5f / (float)BAT) * s;
}

extern "C" void kernel_launch(void* const* d_in, const int* in_sizes, int n_in,
                              void* d_out, int out_size, void* d_ws, size_t ws_size,
                              hipStream_t stream) {
  const float* hseries  = (const float*)d_in[0];
  const float* eps      = (const float*)d_in[1];
  const float* dropu    = (const float*)d_in[2];
  const float* enc_w_ih = (const float*)d_in[3];
  const float* enc_w_hh = (const float*)d_in[4];
  const float* enc_b_ih = (const float*)d_in[5];
  const float* enc_b_hh = (const float*)d_in[6];
  const float* dec_w_ih = (const float*)d_in[7];
  // d_in[8] = dec_w_hh is mathematically unused: decoder h0 == 0 -> gh = b_hh
  const float* dec_b_ih = (const float*)d_in[9];
  const float* dec_b_hh = (const float*)d_in[10];
  const float* w_mu     = (const float*)d_in[11];
  const float* b_mu     = (const float*)d_in[12];
  const float* w_ls     = (const float*)d_in[13];
  const float* b_ls     = (const float*)d_in[14];
  const float* w_pred   = (const float*)d_in[15];
  const float* b_pred   = (const float*)d_in[16];

  // workspace: packed bf16 weights + per-block KL partials
  __bf16* pw_enc_ih = (__bf16*)d_ws;                     // 48*8*32*16  = 196608
  __bf16* pw_enc_hh = pw_enc_ih + 196608;                // 196608
  __bf16* pw_mu     = pw_enc_hh + 196608;                // 16*16*32*16 = 131072
  __bf16* pw_ls     = pw_mu + 131072;                    // 131072
  __bf16* pw_dec    = pw_ls + 131072;                    // 48*16*32*16 = 393216
  float*  klPart    = (float*)(pw_dec + 393216);         // NBLK floats

  {
    int tot = 48 * 8 * 32;
    pack_w<<<(tot + 255) / 256, 256, 0, stream>>>(enc_w_ih, pw_enc_ih, 8, 256, tot);
    pack_w<<<(tot + 255) / 256, 256, 0, stream>>>(enc_w_hh, pw_enc_hh, 8, 256, tot);
  }
  {
    int tot = 16 * 16 * 32;
    pack_w<<<(tot + 255) / 256, 256, 0, stream>>>(w_mu, pw_mu, 16, 512, tot);
    pack_w<<<(tot + 255) / 256, 256, 0, stream>>>(w_ls, pw_ls, 16, 512, tot);
  }
  {
    int tot = 48 * 16 * 32;
    pack_w<<<(tot + 255) / 256, 256, 0, stream>>>(dec_w_ih, pw_dec, 16, 512, tot);
  }

  float* out_pred = (float*)d_out;                          // B*T
  float* out_z    = out_pred + (size_t)BAT * DAYS;          // B*T*L
  float* out_kl   = out_z + (size_t)BAT * DAYS * LAT;       // scalar

  // dynamic LDS: 10*4096 bf16 tiles + pred/kl reduce buffers
  size_t shmem = 10 * 4096 * sizeof(__bf16) + (4 * 32 + 4) * sizeof(float);
  drg_main<<<NBLK, 128, shmem, stream>>>(
      hseries, eps, dropu, enc_b_ih, enc_b_hh, dec_b_ih, dec_b_hh,
      b_mu, b_ls, w_pred, b_pred,
      pw_enc_ih, pw_enc_hh, pw_mu, pw_ls, pw_dec,
      out_pred, out_z, klPart);

  kl_fin<<<1, 1, 0, stream>>>(klPart, out_kl);
}
